// RoPEAttention_40870908788925
// MI455X (gfx1250) — matrix operand order repaired and checked by reading, not compiled, not run on hardware
//
#include <hip/hip_runtime.h>
#include <hip/hip_bf16.h>

typedef __bf16 bf16;
typedef __attribute__((ext_vector_type(16))) __bf16 v16bf;
typedef __attribute__((ext_vector_type(8)))  float  v8f;

#define TSEQ 4096
#define NHEAD 16
#define HDIM 64
#define CDIM 1024
#define MROWS 8192   // B*T

// ---- CDNA5 async global->LDS copy (ASYNCcnt path) -------------------------
#if defined(__AMDGCN__) && __has_builtin(__builtin_amdgcn_global_load_async_to_lds_b128) && __has_builtin(__builtin_amdgcn_s_wait_asynccnt)
#define USE_ASYNC_LDS 1
#else
#define USE_ASYNC_LDS 0
#endif

#if USE_ASYNC_LDS
typedef int v4i_ __attribute__((vector_size(16)));
typedef __attribute__((address_space(1))) v4i_ gv4i_t;  // global
typedef __attribute__((address_space(3))) v4i_ lv4i_t;  // LDS
#endif

__device__ __forceinline__ void copy16(bf16* dstLds, const bf16* srcGlobal) {
#if USE_ASYNC_LDS
  __builtin_amdgcn_global_load_async_to_lds_b128(
      (gv4i_t*)srcGlobal, (lv4i_t*)dstLds, 0, 0);
#else
  *reinterpret_cast<uint4*>(dstLds) = *reinterpret_cast<const uint4*>(srcGlobal);
#endif
}

__device__ __forceinline__ void wait_async() {
#if USE_ASYNC_LDS
  __builtin_amdgcn_s_wait_asynccnt(0);
#endif
}

// ---- WMMA helpers ---------------------------------------------------------
__device__ __forceinline__ v8f wmma_bf16(v16bf a, v16bf b, v8f c) {
  return __builtin_amdgcn_wmma_f32_16x16x32_bf16(false, a, false, b, (short)0, c,
                                                 false, false);
}

// A-matrix 16x32 bf16 fragment from LDS tile stored [row][k], pitch ldw.
__device__ __forceinline__ v16bf frag_a(const bf16* base, int ldw, int row0,
                                        int kBase, int lane) {
  const int m = lane & 15;
  const int kh = (lane >> 4) * 16;
  v16bf a;
#pragma unroll
  for (int v = 0; v < 8; ++v) {
    int k = kBase + 2 * v + kh;
    const bf16* p = base + (row0 + m) * ldw + k;
    a[2 * v]     = p[0];
    a[2 * v + 1] = p[1];
  }
  return a;
}

// B-matrix 32x16 (KxN) fragment where LDS tile is stored transposed [n][k].
__device__ __forceinline__ v16bf frag_b_nk(const bf16* base, int ldw, int n0,
                                           int k0, int lane) {
  const int n = n0 + (lane & 15);
  const int kh = k0 + ((lane >> 4) ? 16 : 0);
  v16bf b;
#pragma unroll
  for (int v = 0; v < 8; ++v) {
    const bf16* p = base + n * ldw + kh + 2 * v;
    b[2 * v]     = p[0];
    b[2 * v + 1] = p[1];
  }
  return b;
}

// B-matrix 32x16 (KxN) fragment where LDS tile is stored natural [k][n].
__device__ __forceinline__ v16bf frag_b_kn(const bf16* base, int ldw, int k0,
                                           int n0, int lane) {
  const int n = n0 + (lane & 15);
  const int kh = k0 + ((lane >> 4) ? 16 : 0);
  v16bf b;
#pragma unroll
  for (int v = 0; v < 8; ++v) {
    b[2 * v]     = base[(kh + 2 * v) * ldw + n];
    b[2 * v + 1] = base[(kh + 2 * v + 1) * ldw + n];
  }
  return b;
}

__global__ void f32_to_bf16_kernel(const float* __restrict__ in,
                                   bf16* __restrict__ out, int n) {
  int i = blockIdx.x * blockDim.x + threadIdx.x;
  int stride = gridDim.x * blockDim.x;
  for (; i < n; i += stride) out[i] = (bf16)in[i];
}

// GEMM: out = A[M x 1024] * W[1024 x 1024]^T  (W row-major [n][k])
// mode 0: store f32 row-major; mode 1: RoPE -> bf16 [B,H,T,D]; mode 2: bf16 [B,H,T,D]
// grid = (16, 64), block = 256 (8 waves); BM=128, BN=64, BK=32, double-buffered.
__global__ __launch_bounds__(256) void gemm_rope_kernel(
    const bf16* __restrict__ A, const bf16* __restrict__ W,
    bf16* __restrict__ outB, float* __restrict__ outF, int mode) {
  const int tid = threadIdx.x;
  const int lane = tid & 31;
  const int w = tid >> 5;         // wave 0..7
  const int wr = w >> 1;          // 0..3 -> 32-row strip
  const int wc = w & 1;           // 0..1 -> 32-col strip
  const int mBase = blockIdx.y * 128;
  const int nBase = blockIdx.x * 64;
  const int rowOfs = (lane >> 4) * 8;
  const int col = lane & 15;

  __shared__ bf16 As[2][128 * 40];  // [row][k], pitch 40 (16B-aligned rows)
  __shared__ bf16 Wt[2][64 * 40];   // [n][k]
  __shared__ float Cs[128 * 66];    // f32 epilogue staging

  v8f acc[2][2];
  {
    v8f z = {};
    acc[0][0] = z; acc[0][1] = z; acc[1][0] = z; acc[1][1] = z;
  }

  auto stage = [&](int buf, int kt) {
    const int k0 = kt * 32;
    for (int idx = tid; idx < 512; idx += 256) {      // A: 128x32
      int r = idx >> 2, cv = (idx & 3) * 8;
      copy16(&As[buf][r * 40 + cv], A + (size_t)(mBase + r) * CDIM + k0 + cv);
    }
    {                                                 // W: 64x32
      int r = tid >> 2, cv = (tid & 3) * 8;
      copy16(&Wt[buf][r * 40 + cv], W + (size_t)(nBase + r) * CDIM + k0 + cv);
    }
  };

  int cur = 0;
  stage(0, 0);
  wait_async();
  __syncthreads();

  for (int kt = 0; kt < 32; ++kt) {
    if (kt + 1 < 32) stage(cur ^ 1, kt + 1);   // overlap next stage with compute

    const bf16* as = As[cur];
    const bf16* wt = Wt[cur];
    v16bf af0 = frag_a(as, 40, wr * 32, 0, lane);
    v16bf af1 = frag_a(as, 40, wr * 32 + 16, 0, lane);
    v16bf bf0 = frag_b_nk(wt, 40, wc * 32, 0, lane);
    v16bf bf1 = frag_b_nk(wt, 40, wc * 32 + 16, 0, lane);
    acc[0][0] = wmma_bf16(af0, bf0, acc[0][0]);
    acc[0][1] = wmma_bf16(af0, bf1, acc[0][1]);
    acc[1][0] = wmma_bf16(af1, bf0, acc[1][0]);
    acc[1][1] = wmma_bf16(af1, bf1, acc[1][1]);

    wait_async();
    __syncthreads();
    cur ^= 1;
  }

  if (mode == 0) {
#pragma unroll
    for (int i = 0; i < 2; ++i)
#pragma unroll
      for (int j = 0; j < 2; ++j) {
        int m0 = mBase + wr * 32 + i * 16 + rowOfs;
        int n0 = nBase + wc * 32 + j * 16 + col;
#pragma unroll
        for (int v = 0; v < 8; ++v)
          outF[(size_t)(m0 + v) * CDIM + n0] = acc[i][j][v];
      }
    return;
  }

  // stage f32 result in LDS so RoPE can pair columns d and d+32
#pragma unroll
  for (int i = 0; i < 2; ++i)
#pragma unroll
    for (int j = 0; j < 2; ++j) {
      int m0 = wr * 32 + i * 16 + rowOfs;
      int n0 = wc * 32 + j * 16 + col;
#pragma unroll
      for (int v = 0; v < 8; ++v) Cs[(m0 + v) * 66 + n0] = acc[i][j][v];
    }
  __syncthreads();

  const int h = blockIdx.x;  // BN == HDIM, so N-tile == head
  if (mode == 1) {
    for (int idx = tid; idx < 128 * 32; idx += 256) {
      int r = idx >> 5;  // local row
      int j = idx & 31;  // rotary pair index
      int gm = mBase + r;
      int b = gm >> 12;
      int t = gm & (TSEQ - 1);
      float q1 = Cs[r * 66 + j];
      float q2 = Cs[r * 66 + j + 32];
      float inv = __expf(-(float)j * 0.2878231366f);  // 10000^(-j/32)
      float ang = (float)t * inv;
      float sn, cn;
      __sincosf(ang, &sn, &cn);
      bf16* dst = outB + (((size_t)(b * NHEAD + h) * TSEQ + t) * HDIM);
      dst[j]      = (bf16)(q1 * cn - q2 * sn);
      dst[j + 32] = (bf16)(q2 * cn + q1 * sn);
    }
  } else {
    for (int idx = tid; idx < 128 * 64; idx += 256) {
      int r = idx >> 6, d = idx & 63;
      int gm = mBase + r;
      int b = gm >> 12;
      int t = gm & (TSEQ - 1);
      outB[(((size_t)(b * NHEAD + h) * TSEQ + t) * HDIM) + d] = (bf16)Cs[r * 66 + d];
    }
  }
}

// Flash attention, causal. grid = (T/64, H, B), block = 128 (4 waves x 16 q-rows).
__global__ __launch_bounds__(128) void flash_attn_kernel(
    const bf16* __restrict__ Q, const bf16* __restrict__ K,
    const bf16* __restrict__ V, bf16* __restrict__ Oattn) {
  const int tid = threadIdx.x;
  const int lane = tid & 31;
  const int w = tid >> 5;
  const int qb = blockIdx.x;
  const int h = blockIdx.y;
  const int b = blockIdx.z;
  const int rowOfs = (lane >> 4) * 8;
  const int col = lane & 15;
  const int qrow0 = qb * 64 + w * 16;

  const size_t headOfs = (size_t)(b * NHEAD + h) * TSEQ * HDIM;
  const bf16* Qh = Q + headOfs;
  const bf16* Kh = K + headOfs;
  const bf16* Vh = V + headOfs;

  __shared__ bf16 Qs[64 * 72];
  __shared__ bf16 Ks[2][64 * 72];
  __shared__ bf16 Vs[2][64 * 72];
  __shared__ bf16 Ps[64 * 72];

  auto stage_kv = [&](int buf, int kb) {
    for (int idx = tid; idx < 512; idx += 128) {
      int r = idx >> 3, cv = (idx & 7) * 8;
      copy16(&Ks[buf][r * 72 + cv], Kh + (size_t)(kb * 64 + r) * HDIM + cv);
      copy16(&Vs[buf][r * 72 + cv], Vh + (size_t)(kb * 64 + r) * HDIM + cv);
    }
  };

  for (int idx = tid; idx < 512; idx += 128) {
    int r = idx >> 3, cv = (idx & 7) * 8;
    copy16(&Qs[r * 72 + cv], Qh + (size_t)(qb * 64 + r) * HDIM + cv);
  }
  stage_kv(0, 0);
  wait_async();
  __syncthreads();

  v16bf aQ0 = frag_a(Qs, 72, w * 16, 0, lane);
  v16bf aQ1 = frag_a(Qs, 72, w * 16, 32, lane);

  v8f acc[4];
  {
    v8f z = {};
    acc[0] = z; acc[1] = z; acc[2] = z; acc[3] = z;
  }
  float mrow[8], lrow[8];
#pragma unroll
  for (int v = 0; v < 8; ++v) { mrow[v] = -1e30f; lrow[v] = 0.0f; }

  int cur = 0;
  for (int kb = 0; kb <= qb; ++kb) {
    if (kb < qb) stage_kv(cur ^ 1, kb + 1);   // overlap next K/V with compute

    const bf16* ks = Ks[cur];
    const bf16* vs = Vs[cur];

    // S = Q K^T over D=64
    v8f s[4];
#pragma unroll
    for (int j = 0; j < 4; ++j) {
      v16bf bk0 = frag_b_nk(ks, 72, j * 16, 0, lane);
      v16bf bk1 = frag_b_nk(ks, 72, j * 16, 32, lane);
      v8f t = {};
      t = wmma_bf16(aQ0, bk0, t);
      t = wmma_bf16(aQ1, bk1, t);
      s[j] = t;
    }

    // scale + causal mask
#pragma unroll
    for (int j = 0; j < 4; ++j) {
      int key = kb * 64 + j * 16 + col;
#pragma unroll
      for (int v = 0; v < 8; ++v) {
        float val = s[j][v] * 0.125f;
        int qi = qrow0 + rowOfs + v;
        if (key > qi) val = -1e30f;
        s[j][v] = val;
      }
    }

    // online softmax (rows = (VGPR v, 16-lane half); reduce across 16 lanes)
#pragma unroll
    for (int v = 0; v < 8; ++v) {
      float rm = -1e30f;
#pragma unroll
      for (int j = 0; j < 4; ++j) rm = fmaxf(rm, s[j][v]);
#pragma unroll
      for (int off = 1; off < 16; off <<= 1)
        rm = fmaxf(rm, __shfl_xor(rm, off, 32));
      float mnew = fmaxf(mrow[v], rm);
      float corr = __expf(mrow[v] - mnew);
      float rsum = 0.0f;
#pragma unroll
      for (int j = 0; j < 4; ++j) {
        float p = __expf(s[j][v] - mnew);
        s[j][v] = p;
        rsum += p;
      }
#pragma unroll
      for (int off = 1; off < 16; off <<= 1) rsum += __shfl_xor(rsum, off, 32);
      lrow[v] = lrow[v] * corr + rsum;
      mrow[v] = mnew;
#pragma unroll
      for (int j = 0; j < 4; ++j) acc[j][v] *= corr;
    }

    // restage P (bf16) through this wave's private LDS strip -> A layout
#pragma unroll
    for (int j = 0; j < 4; ++j)
#pragma unroll
      for (int v = 0; v < 8; ++v)
        Ps[(w * 16 + rowOfs + v) * 72 + j * 16 + col] = (bf16)s[j][v];
    v16bf aP0 = frag_a(Ps, 72, w * 16, 0, lane);
    v16bf aP1 = frag_a(Ps, 72, w * 16, 32, lane);

    // O += P V
#pragma unroll
    for (int j = 0; j < 4; ++j) {
      v16bf bv0 = frag_b_kn(vs, 72, 0, j * 16, lane);
      v16bf bv1 = frag_b_kn(vs, 72, 32, j * 16, lane);
      acc[j] = wmma_bf16(aP0, bv0, acc[j]);
      acc[j] = wmma_bf16(aP1, bv1, acc[j]);
    }

    wait_async();
    __syncthreads();
    cur ^= 1;
  }

  // normalize and store to [B, T, C] bf16 for the final projection
#pragma unroll
  for (int v = 0; v < 8; ++v) {
    float invl = 1.0f / lrow[v];
    int t = qrow0 + rowOfs + v;
#pragma unroll
    for (int j = 0; j < 4; ++j)
      Oattn[(size_t)(b * TSEQ + t) * CDIM + h * HDIM + j * 16 + col] =
          (bf16)(acc[j][v] * invl);
  }
}

extern "C" void kernel_launch(void* const* d_in, const int* in_sizes, int n_in,
                              void* d_out, int out_size, void* d_ws, size_t ws_size,
                              hipStream_t stream) {
  (void)in_sizes; (void)n_in; (void)out_size; (void)ws_size;
  const float* x  = (const float*)d_in[0];
  const float* Wq = (const float*)d_in[1];
  const float* Wk = (const float*)d_in[2];
  const float* Wv = (const float*)d_in[3];
  const float* Wo = (const float*)d_in[4];
  float* out = (float*)d_out;

  const size_t MK = (size_t)MROWS * CDIM;
  const size_t WW = (size_t)CDIM * CDIM;
  char* ws = (char*)d_ws;
  bf16* xb   = (bf16*)ws; ws += MK * 2;
  bf16* wqb  = (bf16*)ws; ws += WW * 2;
  bf16* wkb  = (bf16*)ws; ws += WW * 2;
  bf16* wvb  = (bf16*)ws; ws += WW * 2;
  bf16* wob  = (bf16*)ws; ws += WW * 2;
  bf16* q    = (bf16*)ws; ws += MK * 2;
  bf16* k    = (bf16*)ws; ws += MK * 2;
  bf16* v    = (bf16*)ws; ws += MK * 2;
  bf16* attn = (bf16*)ws; ws += MK * 2;

  f32_to_bf16_kernel<<<2048, 256, 0, stream>>>(x, xb, (int)MK);
  f32_to_bf16_kernel<<<512, 256, 0, stream>>>(Wq, wqb, (int)WW);
  f32_to_bf16_kernel<<<512, 256, 0, stream>>>(Wk, wkb, (int)WW);
  f32_to_bf16_kernel<<<512, 256, 0, stream>>>(Wv, wvb, (int)WW);
  f32_to_bf16_kernel<<<512, 256, 0, stream>>>(Wo, wob, (int)WW);

  dim3 gGemm(CDIM / 64, MROWS / 128);  // (16, 64)
  gemm_rope_kernel<<<gGemm, 256, 0, stream>>>(xb, wqb, q, nullptr, 1);
  gemm_rope_kernel<<<gGemm, 256, 0, stream>>>(xb, wkb, k, nullptr, 1);
  gemm_rope_kernel<<<gGemm, 256, 0, stream>>>(xb, wvb, v, nullptr, 2);

  flash_attn_kernel<<<dim3(TSEQ / 64, NHEAD, 2), 128, 0, stream>>>(q, k, v, attn);

  gemm_rope_kernel<<<gGemm, 256, 0, stream>>>(attn, wob, nullptr, out, 0);
}
